// QCNNLSTM_13537736917323
// MI455X (gfx1250) — matrix operand order repaired and checked by reading, not compiled, not run on hardware
//
#include <hip/hip_runtime.h>
#include <hip/hip_fp16.h>
#include <math.h>
#include <stdint.h>

typedef _Float16 h16;
typedef __attribute__((ext_vector_type(16))) _Float16 v16h;
typedef __attribute__((ext_vector_type(8)))  float    v8f;
typedef __attribute__((ext_vector_type(4)))  unsigned uint32x4;
typedef __attribute__((ext_vector_type(8)))  int      int32x8;
typedef __attribute__((ext_vector_type(4)))  int      int32x4;

#define DEVINL __device__ __forceinline__

DEVINL float uasf(unsigned u) { return __uint_as_float(u); }
DEVINL unsigned fasu(float f) { return __float_as_uint(f); }

// symmetric fake-quant value: clip(round(x/s), lo, hi)*s
DEVINL float qclip(float x, float s, float lo, float hi) {
    return fminf(fmaxf(rintf(x / s), lo), hi) * s;
}
// fq_unit helpers
DEVINL float qsig63(float x) {        // fq_unit(sigmoid, 6, unsigned)
    float sg = 1.f / (1.f + __expf(-x));
    return rintf(fminf(fmaxf(sg, 0.f), 1.f) * 63.f) * (1.f / 63.f);
}
DEVINL float qtanh31(float x) {       // fq_unit(tanh, 6, signed)
    return rintf(fminf(fmaxf(tanhf(x), -1.f), 1.f) * 31.f) * (1.f / 31.f);
}

// ---------------- Tensor Data Mover: contiguous 1-D global->LDS copy ----------------
// D# per CDNA5 ISA §8.3/8.4: group0 {count=1, lds_addr, global_addr, type=2},
// group1 {data_size=3 (8B units), tensor_dim0=tile_dim0=n/8, tensor_dim1=1,
//         dim0/dim1 strides = n/8}. Issued by one wave; tracked with TENSORcnt.
DEVINL void tdm_load_1d(void* ldsDst, const void* gsrc, unsigned nbytes) {
    unsigned n8 = nbytes >> 3;                       // 8-byte elements (nbytes % 8 == 0)
    unsigned lds = (unsigned)(uintptr_t)ldsDst;      // generic LDS addr: offset in [31:0]
    unsigned long long ga = (unsigned long long)(uintptr_t)gsrc;
    uint32x4 g0 = { 1u,                              // count=1 (valid), no gather
                    lds,                             // lds_addr
                    (unsigned)ga,                    // global_addr[31:0]
                    ((unsigned)(ga >> 32) & 0x01ffffffu) | 0x80000000u };  // ga[56:32] | type=2
    int32x8 g1 = { (int)0x00030000,                  // workgroup_mask=0, data_size=3 (8B)
                   (int)(n8 << 16),                  // tensor_dim0[15:0] << 16
                   (int)((n8 >> 16) | (1u << 16)),   // tensor_dim0[31:16] | tensor_dim1=1
                   (int)((n8 & 0xffffu) << 16),      // tile_dim0 (16-bit)
                   0,                                // tile_dim1=0 (unused), tile_dim2=0
                   (int)n8,                          // tensor_dim0_stride[31:0]
                   (int)((n8 & 0xffffu) << 16),      // dim0_stride[47:32]=0 | dim1_stride[15:0]
                   0 };                              // dim1_stride[47:16]=0
    int32x4 z4 = {0, 0, 0, 0};
#if __has_include(<hip/amd_detail/amd_gfx1250_TDM.h>)
    int32x8 z8 = {0, 0, 0, 0, 0, 0, 0, 0};
    __builtin_amdgcn_tensor_load_to_lds(g0, g1, z4, z4, z8, 0);
#else
    __builtin_amdgcn_tensor_load_to_lds(g0, g1, z4, z4, 0);
#endif
}

// ---------------- WMMA fragment loaders (wave32, 16x16x32 f16) ----------------
// A: 16x32 tile from row-major [rows][lda] f16.  Lane l: M = l&15.
// lanes 0-15: K=0..15 contiguous; lanes 16-31: K=16..31.
DEVINL v16h ldsA(const h16* __restrict__ A, int lda, int m0, int k0, int lane) {
    const h16* p = A + (size_t)(m0 + (lane & 15)) * lda + k0 + ((lane >> 4) << 4);
    v16h a;
#pragma unroll
    for (int i = 0; i < 16; ++i) a[i] = p[i];
    return a;
}
// B: 32x16 tile; source stored as [N][K] row-major (i.e. weight rows).
// Lane l: N = l&15, contiguous K run of 16 starting at k0 + (l>=16)*16.
DEVINL v16h ldsB(const h16* __restrict__ B, int ldb, int n0, int k0, int lane) {
    const h16* p = B + (size_t)(n0 + (lane & 15)) * ldb + k0 + ((lane >> 4) << 4);
    v16h b;
#pragma unroll
    for (int i = 0; i < 16; ++i) b[i] = p[i];
    return b;
}
// C/D mapping: VGPR v -> row m0 + v + 8*(lane>=16), col n0 + (lane&15)

// ---------------- small utility kernels ----------------
__global__ void kInit(unsigned* sc) {
    if (threadIdx.x < 64) sc[threadIdx.x] = 0u;
}

__global__ void kMaxAbs(const float* __restrict__ src, int n, unsigned* slot) {
    __shared__ unsigned redU;
    if (threadIdx.x == 0) redU = 0u;
    __syncthreads();
    float lm = 0.f;
    for (int i = blockIdx.x * blockDim.x + threadIdx.x; i < n; i += gridDim.x * blockDim.x)
        lm = fmaxf(lm, fabsf(src[i]));
    atomicMax(&redU, fasu(lm));
    __syncthreads();
    if (threadIdx.x == 0) atomicMax(slot, redU);
}

// one block: weight-tensor scale = max(max|w|,1e-8)/qmax, stored as float
__global__ void kWScale(const float* __restrict__ src, int n, float qmax, float* slotF) {
    __shared__ unsigned redU;
    if (threadIdx.x == 0) redU = 0u;
    __syncthreads();
    float lm = 0.f;
    for (int i = threadIdx.x; i < n; i += blockDim.x) lm = fmaxf(lm, fabsf(src[i]));
    atomicMax(&redU, fasu(lm));
    __syncthreads();
    if (threadIdx.x == 0) slotF[0] = fmaxf(uasf(redU), 1e-8f) / qmax;
}

// quantize weights -> f16, per-group scale (group = perSize consecutive elems)
__global__ void kQuantGroup(const float* __restrict__ src, h16* __restrict__ dst, int n,
                            int perSize, const float* __restrict__ scf, int slotBase,
                            float lo, float hi) {
    int idx = blockIdx.x * 256 + threadIdx.x;
    if (idx >= n) return;
    float s = scf[slotBase + idx / perSize];
    dst[idx] = (h16)qclip(src[idx], s, lo, hi);
}

// conv2 weights [O=128][C=64][K=3] -> f16 [O][k=dk*64+c] (im2col order), 4-bit quant
__global__ void kQuantW2(const float* __restrict__ w, h16* __restrict__ dst,
                         const float* __restrict__ scf) {
    int idx = blockIdx.x * 256 + threadIdx.x;
    if (idx >= 24576) return;
    float s = scf[17];
    int o = idx / 192, kk = idx - o * 192;
    int dk = kk >> 6, c = kk & 63;
    dst[idx] = (h16)qclip(w[(o * 64 + c) * 3 + dk], s, -8.f, 7.f);
}

// ---------------- conv1 (tiny: 1->64 ch, k=3) direct VALU ----------------
__global__ void kConv1(const float* __restrict__ x, const float* __restrict__ w1,
                       const float* __restrict__ b1, unsigned* __restrict__ sc,
                       float* __restrict__ outA) {
    __shared__ float wq[192];
    __shared__ unsigned redU;
    int tid = threadIdx.x;
    const float* scf = (const float*)sc;
    float sw = scf[16];
    float sx = fmaxf(uasf(sc[0]), 1e-8f) / 127.f;     // fq(x,8)
    if (tid < 192) wq[tid] = qclip(w1[tid], sw, -8.f, 7.f);
    if (tid == 0) redU = 0u;
    __syncthreads();
    int idx = blockIdx.x * 256 + tid;                 // b*512 + l
    int b = idx >> 9, l = idx & 511;
    const float* xb = x + (size_t)b * 512;
    float x0 = qclip(xb[l], sx, -128.f, 127.f);
    float xm = (l > 0)   ? qclip(xb[l - 1], sx, -128.f, 127.f) : 0.f;
    float xp = (l < 511) ? qclip(xb[l + 1], sx, -128.f, 127.f) : 0.f;
    float lm = 0.f;
    float* ob = outA + (size_t)b * 64 * 512 + l;
#pragma unroll 4
    for (int o = 0; o < 64; ++o) {
        float z = xm * wq[o * 3] + x0 * wq[o * 3 + 1] + xp * wq[o * 3 + 2] + b1[o];
        float r = fmaxf(z, 0.f);
        ob[(size_t)o * 512] = r;
        lm = fmaxf(lm, r);
    }
    atomicMax(&redU, fasu(lm));
    __syncthreads();
    if (tid == 0) atomicMax(&sc[1], redU);
}

// quant4u(relu1) + bn1 -> f16  (conv2 input)
__global__ void kPw1(const float* __restrict__ A, const unsigned* __restrict__ sc,
                     const float* __restrict__ g, const float* __restrict__ bb,
                     const float* __restrict__ mm, const float* __restrict__ vv,
                     h16* __restrict__ outB, int n) {
    int idx = blockIdx.x * 256 + threadIdx.x;
    if (idx >= n) return;
    float s = fmaxf(uasf(sc[1]), 1e-8f) / 15.f;
    int c = (idx >> 9) & 63;
    float q = qclip(A[idx], s, 0.f, 15.f);
    float inv = g[c] * rsqrtf(vv[c] + 1e-5f);
    outB[idx] = (h16)((q - mm[c]) * inv + bb[c]);
}

// ---------------- conv2 as implicit GEMM (WMMA): M=B*L, K=192, N=128 ----------------
__global__ void kConv2(const h16* __restrict__ Bq, const h16* __restrict__ w2q,
                       const float* __restrict__ b2, unsigned* __restrict__ sc,
                       float* __restrict__ Cg) {
    extern __shared__ char smem[];
    h16* As = (h16*)smem;                 // [128][192]
    h16* Bs = As + 128 * 192;             // [128][192]
    unsigned* redU = (unsigned*)(Bs + 128 * 192);
    int tid = threadIdx.x;
    int m0g = blockIdx.x * 128;
    int b = m0g >> 9, l0 = m0g & 511;
    // weights: TDM async copy by wave 0, overlapped with manual im2col below
    if (tid < 32) tdm_load_1d(Bs, w2q, 128 * 192 * 2);
    {   // im2col A tile with zero 'same' padding
        const h16* yb = Bq + (size_t)b * 64 * 512;
        for (int e = tid; e < 128 * 192; e += 256) {
            int m = e / 192, kk = e - m * 192;
            int dk = kk >> 6, c = kk & 63;
            int pos = l0 + m + dk - 1;
            h16 val = (h16)0.f;
            if (pos >= 0 && pos < 512) val = yb[(size_t)c * 512 + pos];
            As[m * 192 + kk] = val;
        }
    }
    if (tid == 0) *redU = 0u;
    if (tid < 32) __builtin_amdgcn_s_wait_tensorcnt(0);
    __syncthreads();
    int w = tid >> 5, lane = tid & 31;
    float lm = 0.f;
    for (int i = 0; i < 8; ++i) {
        int tile = w * 8 + i;
        int m0 = (tile >> 3) * 16, n0 = (tile & 7) * 16;
        v8f acc = {};
#pragma unroll
        for (int kb = 0; kb < 6; ++kb) {
            v16h a = ldsA(As, 192, m0, kb * 32, lane);
            v16h bf = ldsB(Bs, 192, n0, kb * 32, lane);
            acc = __builtin_amdgcn_wmma_f32_16x16x32_f16(false, a, false, bf, (short)0,
                                                         acc, false, false);
        }
        int hi = lane >> 4, col = lane & 15;
        int n = n0 + col;
        float bias = b2[n];
#pragma unroll
        for (int vv = 0; vv < 8; ++vv) {
            int mrow = m0 + (hi << 3) + vv;
            float r = fmaxf(acc[vv] + bias, 0.f);
            Cg[((size_t)(b * 128 + n)) * 512 + l0 + mrow] = r;
            lm = fmaxf(lm, r);
        }
    }
    atomicMax(redU, fasu(lm));
    __syncthreads();
    if (tid == 0) atomicMax(&sc[2], *redU);
}

// quant4u(relu2)+bn2+maxpool2+transpose -> D [b][t][c] f32 ; track max|.|
__global__ void kPw2(const float* __restrict__ C, unsigned* __restrict__ sc,
                     const float* __restrict__ g, const float* __restrict__ bb,
                     const float* __restrict__ mm, const float* __restrict__ vv,
                     float* __restrict__ D, int n) {
    __shared__ unsigned redU;
    if (threadIdx.x == 0) redU = 0u;
    __syncthreads();
    int idx = blockIdx.x * 256 + threadIdx.x;
    float lm = 0.f;
    if (idx < n) {
        float s = fmaxf(uasf(sc[2]), 1e-8f) / 15.f;
        int c = idx & 127, t = (idx >> 7) & 255, b = idx >> 15;
        size_t base = ((size_t)(b * 128 + c)) * 512 + 2 * t;
        float q0 = qclip(C[base], s, 0.f, 15.f);
        float q1 = qclip(C[base + 1], s, 0.f, 15.f);
        float inv = g[c] * rsqrtf(vv[c] + 1e-5f);
        float z0 = (q0 - mm[c]) * inv + bb[c];
        float z1 = (q1 - mm[c]) * inv + bb[c];
        float d = fmaxf(z0, z1);
        D[idx] = d;
        lm = fabsf(d);
    }
    atomicMax(&redU, fasu(lm));
    __syncthreads();
    if (threadIdx.x == 0) atomicMax(&sc[3], redU);
}

// fq(seq,8) -> f16
__global__ void kPw3(const float* __restrict__ D, const unsigned* __restrict__ sc,
                     h16* __restrict__ E, int n) {
    int idx = blockIdx.x * 256 + threadIdx.x;
    if (idx >= n) return;
    float s = fmaxf(uasf(sc[3]), 1e-8f) / 127.f;
    E[idx] = (h16)qclip(D[idx], s, -128.f, 127.f);
}

// ---------------- x-projection GEMM (WMMA): [B*T,128] x wi^T[128,512] + bias ----------------
__global__ void kXproj(const h16* __restrict__ E, const h16* __restrict__ wiq,
                       const float* __restrict__ bih, const float* __restrict__ bhh,
                       float* __restrict__ xp) {
    extern __shared__ char smem[];
    h16* As = (h16*)smem;            // [128][128]
    h16* Bs = As + 128 * 128;        // [128][128]
    int tid = threadIdx.x;
    int m0g = blockIdx.x * 128, n0g = blockIdx.y * 128;
    if (tid < 32) {                  // wave 0 drives TDM for both operand tiles
        tdm_load_1d(As, E + (size_t)m0g * 128, 128 * 128 * 2);
        tdm_load_1d(Bs, wiq + (size_t)n0g * 128, 128 * 128 * 2);
        __builtin_amdgcn_s_wait_tensorcnt(0);
    }
    __syncthreads();
    int w = tid >> 5, lane = tid & 31;
    for (int i = 0; i < 8; ++i) {
        int tile = w * 8 + i;
        int m0 = (tile >> 3) * 16, n0 = (tile & 7) * 16;
        v8f acc = {};
#pragma unroll
        for (int kb = 0; kb < 4; ++kb) {
            v16h a = ldsA(As, 128, m0, kb * 32, lane);
            v16h bf = ldsB(Bs, 128, n0, kb * 32, lane);
            acc = __builtin_amdgcn_wmma_f32_16x16x32_f16(false, a, false, bf, (short)0,
                                                         acc, false, false);
        }
        int hi = lane >> 4, col = lane & 15;
        int ng = n0g + n0 + col;
        float bias = bih[ng] + bhh[ng];
#pragma unroll
        for (int vv = 0; vv < 8; ++vv) {
            int mg = m0g + m0 + (hi << 3) + vv;       // row = b*256 + t
            int b = mg >> 8, t = mg & 255;
            xp[((size_t)t * 256 + b) * 512 + ng] = acc[vv] + bias;   // store [t][b][n]
        }
    }
}

// ---------------- persistent single-workgroup LSTM ----------------
// 1024 threads (32 waves) on one WGP. h (f16) + quantized w_hh (f16, TDM preload)
// in LDS, cell state in registers (32 cells/thread). Per-step per-tensor quant
// maxima via LDS atomic-max between workgroup barriers. WMMA recurrent GEMM.
__global__ void __launch_bounds__(1024) kLstm(const h16* __restrict__ whqg,
                                              const float* __restrict__ xp,
                                              float* __restrict__ accg,
                                              h16* __restrict__ hnq) {
    extern __shared__ char smem[];
    h16* whs = (h16*)smem;                 // [512][128] quantized w_hh (row n, col k)
    h16* hs = whs + 512 * 128;             // [256][128] hidden state f16
    unsigned* red = (unsigned*)(hs + 256 * 128);  // [3] step maxima
    int tid = threadIdx.x;
    if (tid < 32) tdm_load_1d(whs, whqg, 512 * 128 * 2);   // 128 KB via TDM
    {
        uint32_t* hz = (uint32_t*)hs;
        for (int i = tid; i < 16384; i += 1024) hz[i] = 0u;
    }
    if (tid < 32) __builtin_amdgcn_s_wait_tensorcnt(0);
    float cst[32], ogt[32], tmp[32];
#pragma unroll
    for (int k = 0; k < 32; ++k) cst[k] = 0.f;
    int w = tid >> 5, lane = tid & 31;
    int hi = lane >> 4, col = lane & 15;

    for (int t = 0; t < 256; ++t) {
        if (tid < 3) red[tid] = 0u;
        if (t + 1 < 256) {             // prefetch next step's x-projection into GL2/L0
            const float* nx = xp + (size_t)(t + 1) * 131072 + (size_t)tid * 128;
            __builtin_prefetch(nx, 0, 0);
            __builtin_prefetch(nx + 64, 0, 0);
        }
        __syncthreads();
        // ---- phase 1: acc = h @ wh^T + xproj[t] (+bias baked in), max|acc| ----
        const float* xpt = xp + (size_t)t * 256 * 512;
        float lm = 0.f;
        for (int i = 0; i < 16; ++i) {
            int tile = w * 16 + i;
            int m0 = (tile >> 5) * 16, n0 = (tile & 31) * 16;
            v8f acc;
#pragma unroll
            for (int vv = 0; vv < 8; ++vv)
                acc[vv] = xpt[(size_t)(m0 + (hi << 3) + vv) * 512 + n0 + col];
#pragma unroll
            for (int kb = 0; kb < 4; ++kb) {
                v16h a = ldsA(hs, 128, m0, kb * 32, lane);
                v16h bf = ldsB(whs, 128, n0, kb * 32, lane);
                acc = __builtin_amdgcn_wmma_f32_16x16x32_f16(false, a, false, bf, (short)0,
                                                             acc, false, false);
            }
#pragma unroll
            for (int vv = 0; vv < 8; ++vv) {
                float z = acc[vv];
                accg[(size_t)(m0 + (hi << 3) + vv) * 512 + n0 + col] = z;
                lm = fmaxf(lm, fabsf(z));
            }
        }
        atomicMax(&red[0], fasu(lm));
        __threadfence();
        __syncthreads();
        // ---- phase 2: fq(acc,6); gates; c_pre = f*c + i*g; max|c_pre| ----
        float sacc = fmaxf(uasf(red[0]), 1e-8f) / 31.f;
        lm = 0.f;
        volatile const float* av = accg;   // L0-bypassing reads (cross-CU in WGP)
#pragma unroll
        for (int k = 0; k < 32; ++k) {
            int j = tid + (k << 10);
            int b = j >> 7, hh = j & 127;
            size_t base = (size_t)b * 512 + hh;
            float ai = qclip(av[base], sacc, -32.f, 31.f);
            float af = qclip(av[base + 128], sacc, -32.f, 31.f);
            float ag = qclip(av[base + 256], sacc, -32.f, 31.f);
            float ao = qclip(av[base + 384], sacc, -32.f, 31.f);
            float ig = qsig63(ai);
            float fg = qsig63(af);
            ogt[k] = qsig63(ao);
            float gg = qtanh31(ag);
            float cp = fg * cst[k] + ig * gg;
            tmp[k] = cp;
            lm = fmaxf(lm, fabsf(cp));
        }
        atomicMax(&red[1], fasu(lm));
        __syncthreads();
        // ---- phase 3: c = fq(c_pre,6); hpre = o * fq_unit(tanh(c)); max|hpre| ----
        float scc = fmaxf(uasf(red[1]), 1e-8f) / 31.f;
        lm = 0.f;
#pragma unroll
        for (int k = 0; k < 32; ++k) {
            float cc = qclip(tmp[k], scc, -32.f, 31.f);
            cst[k] = cc;
            float hp = ogt[k] * qtanh31(cc);
            tmp[k] = hp;
            lm = fmaxf(lm, fabsf(hp));
        }
        atomicMax(&red[2], fasu(lm));
        __syncthreads();
        // ---- phase 4: h = fq(hpre,8) -> LDS (f16) ----
        float sh = fmaxf(uasf(red[2]), 1e-8f) / 127.f;
#pragma unroll
        for (int k = 0; k < 32; ++k) {
            int j = tid + (k << 10);
            float hv = qclip(tmp[k], sh, -128.f, 127.f);
            hs[j] = (h16)hv;
            if (t == 255) hnq[j] = (h16)hv;
        }
        __syncthreads();
    }
}

// ---------------- generic head GEMM (WMMA): out = relu(A @ B^T + bias), per-head max ----------------
__global__ void kHeadGemm(const h16* __restrict__ Abase, int aStride,
                          const h16* __restrict__ Bbase, int bStride,
                          const float* __restrict__ biasBase, int biasStride,
                          float* __restrict__ outBase, unsigned* __restrict__ sc,
                          int slotBase, int M, int N, int K) {
    extern __shared__ char smem[];
    h16* As = (h16*)smem;            // [128][K]
    h16* Bs = As + 128 * K;          // [128][K]
    unsigned* redU = (unsigned*)(Bs + 128 * K);
    int h = blockIdx.z;
    const h16* A = Abase + (size_t)h * aStride;
    const h16* B = Bbase + (size_t)h * bStride;
    const float* bias = biasBase + (size_t)h * biasStride;
    float* out = outBase + (size_t)h * M * N;
    int nblk = N >> 7;
    int m0g = (blockIdx.x / nblk) * 128, n0g = (blockIdx.x % nblk) * 128;
    int tid = threadIdx.x;
    if (tid < 32) {                  // TDM staging of both operand tiles
        tdm_load_1d(As, A + (size_t)m0g * K, 128 * K * 2);
        tdm_load_1d(Bs, B + (size_t)n0g * K, 128 * K * 2);
    }
    if (tid == 0) *redU = 0u;
    if (tid < 32) __builtin_amdgcn_s_wait_tensorcnt(0);
    __syncthreads();
    int w = tid >> 5, lane = tid & 31;
    float lm = 0.f;
    int nk = K >> 5;
    for (int i = 0; i < 8; ++i) {
        int tile = w * 8 + i;
        int m0 = (tile >> 3) * 16, n0 = (tile & 7) * 16;
        v8f acc = {};
        for (int kb = 0; kb < nk; ++kb) {
            v16h a = ldsA(As, K, m0, kb * 32, lane);
            v16h bf = ldsB(Bs, K, n0, kb * 32, lane);
            acc = __builtin_amdgcn_wmma_f32_16x16x32_f16(false, a, false, bf, (short)0,
                                                         acc, false, false);
        }
        int hi = lane >> 4, col = lane & 15;
        int ng = n0g + n0 + col;
        float bv = bias[ng];
#pragma unroll
        for (int vv = 0; vv < 8; ++vv) {
            int mg = m0g + m0 + (hi << 3) + vv;
            float r = fmaxf(acc[vv] + bv, 0.f);
            out[(size_t)mg * N + ng] = r;
            lm = fmaxf(lm, r);
        }
    }
    atomicMax(redU, fasu(lm));
    __syncthreads();
    if (tid == 0) atomicMax(&sc[slotBase + h], *redU);
}

// per-head quant4u + head BN -> f16
__global__ void kHeadPw(const float* __restrict__ pre, const unsigned* __restrict__ sc,
                        int slotBase, int layer,
                        const float* __restrict__ bg, const float* __restrict__ bb,
                        const float* __restrict__ bm, const float* __restrict__ bv,
                        h16* __restrict__ yq) {
    int idx = blockIdx.x * 256 + threadIdx.x;      // [h][b][p] flat, n = 262144
    int p = idx & 255, h = idx >> 16;
    float s = fmaxf(uasf(sc[slotBase + h]), 1e-8f) / 15.f;
    float q = qclip(pre[idx], s, 0.f, 15.f);
    int bi = (h * 3 + layer) * 256 + p;
    float inv = bg[bi] * rsqrtf(bv[bi] + 1e-5f);
    yq[idx] = (h16)((q - bm[bi]) * inv + bb[bi]);
}

// head output: out[b,h] = dot(y[h,b,:], woq[h,:]) + bo[h]
__global__ void kOut(const h16* __restrict__ yq, const h16* __restrict__ woq,
                     const float* __restrict__ bo, float* __restrict__ out) {
    __shared__ float sh[256];
    int bid = blockIdx.x;          // 0..1023
    int b = bid >> 2, h = bid & 3;
    int tid = threadIdx.x;
    sh[tid] = (float)yq[((size_t)(h * 256 + b)) * 256 + tid] * (float)woq[h * 256 + tid];
    __syncthreads();
    for (int s = 128; s > 0; s >>= 1) {
        if (tid < s) sh[tid] += sh[tid + s];
        __syncthreads();
    }
    if (tid == 0) out[b * 4 + h] = sh[0] + bo[h];
}

// ---------------- host ----------------
extern "C" void kernel_launch(void* const* d_in, const int* in_sizes, int n_in,
                              void* d_out, int out_size, void* d_ws, size_t ws_size,
                              hipStream_t stream) {
    (void)in_sizes; (void)n_in; (void)out_size; (void)ws_size;
    const float* x    = (const float*)d_in[0];
    const float* w1   = (const float*)d_in[1];
    const float* b1   = (const float*)d_in[2];
    const float* bn1g = (const float*)d_in[3];
    const float* bn1b = (const float*)d_in[4];
    const float* bn1m = (const float*)d_in[5];
    const float* bn1v = (const float*)d_in[6];
    const float* w2   = (const float*)d_in[7];
    const float* b2   = (const float*)d_in[8];
    const float* bn2g = (const float*)d_in[9];
    const float* bn2b = (const float*)d_in[10];
    const float* bn2m = (const float*)d_in[11];
    const float* bn2v = (const float*)d_in[12];
    const float* wih  = (const float*)d_in[13];
    const float* whh  = (const float*)d_in[14];
    const float* bih  = (const float*)d_in[15];
    const float* bhh  = (const float*)d_in[16];
    const float* hw0  = (const float*)d_in[17];
    const float* hb0  = (const float*)d_in[18];
    const float* hwh  = (const float*)d_in[19];
    const float* hbh  = (const float*)d_in[20];
    const float* hbg  = (const float*)d_in[21];
    const float* hbb  = (const float*)d_in[22];
    const float* hbm  = (const float*)d_in[23];
    const float* hbv  = (const float*)d_in[24];
    const float* hwo  = (const float*)d_in[25];
    const float* hbo  = (const float*)d_in[26];

    char* ws = (char*)d_ws;
    unsigned* sc = (unsigned*)ws;            // 64 scale/max slots
    float* scf = (float*)ws;
    h16*   w2q   = (h16*)(ws + 4096);        //  49152 B
    h16*   wiq   = (h16*)(ws + 53248);       // 131072 B
    h16*   whq   = (h16*)(ws + 184320);      // 131072 B
    h16*   hw0q  = (h16*)(ws + 315392);      // 262144 B
    h16*   hwhq  = (h16*)(ws + 577536);      // 1048576 B
    h16*   hwoq  = (h16*)(ws + 1626112);     //   2048 B
    float* accg  = (float*)(ws + 1628160);   // 524288 B  [256][512]
    h16*   hnq   = (h16*)(ws + 2152448);     //  65536 B  [256][128]
    float* hpre  = (float*)(ws + 2217984);   // 1048576 B [4][256][256]
    h16*   hyq   = (h16*)(ws + 3266560);     // 524288 B  [4][256][256]
    float* Abuf  = (float*)(ws + 4194304);   // 33.5 MB conv1 relu  (aliased: D seq f32)
    float* Dbuf  = Abuf;
    h16*   Bq    = (h16*)(ws + 37748736);    // 16.8 MB y1q f16     (aliased: E seq f16)
    h16*   Ebuf  = Bq;
    float* Cbuf  = (float*)(ws + 54525952);  // 67 MB conv2 relu
    float* xproj = Cbuf;                     // 134 MB, overlays dead Cbuf

    kInit<<<1, 64, 0, stream>>>(sc);
    kMaxAbs<<<256, 256, 0, stream>>>(x, 131072, &sc[0]);
    // weight scales
    kWScale<<<1, 256, 0, stream>>>(w1, 192, 7.f, &scf[16]);
    kWScale<<<1, 256, 0, stream>>>(w2, 24576, 7.f, &scf[17]);
    kWScale<<<1, 256, 0, stream>>>(wih, 65536, 127.f, &scf[18]);
    kWScale<<<1, 256, 0, stream>>>(whh, 65536, 127.f, &scf[19]);
    for (int h = 0; h < 4; ++h)
        kWScale<<<1, 256, 0, stream>>>(hw0 + h * 32768, 32768, 7.f, &scf[20 + h]);
    for (int hl = 0; hl < 8; ++hl)
        kWScale<<<1, 256, 0, stream>>>(hwh + hl * 65536, 65536, 7.f, &scf[24 + hl]);
    for (int h = 0; h < 4; ++h)
        kWScale<<<1, 256, 0, stream>>>(hwo + h * 256, 256, 7.f, &scf[32 + h]);
    // quantize weights to f16
    kQuantW2<<<96, 256, 0, stream>>>(w2, w2q, scf);
    kQuantGroup<<<256, 256, 0, stream>>>(wih, wiq, 65536, 65536, scf, 18, -128.f, 127.f);
    kQuantGroup<<<256, 256, 0, stream>>>(whh, whq, 65536, 65536, scf, 19, -128.f, 127.f);
    kQuantGroup<<<512, 256, 0, stream>>>(hw0, hw0q, 131072, 32768, scf, 20, -8.f, 7.f);
    kQuantGroup<<<2048, 256, 0, stream>>>(hwh, hwhq, 524288, 65536, scf, 24, -8.f, 7.f);
    kQuantGroup<<<4, 256, 0, stream>>>(hwo, hwoq, 1024, 256, scf, 32, -8.f, 7.f);
    // CNN front-end
    kConv1<<<512, 256, 0, stream>>>(x, w1, b1, sc, Abuf);
    kPw1<<<32768, 256, 0, stream>>>(Abuf, sc, bn1g, bn1b, bn1m, bn1v, Bq, 8388608);
    kConv2<<<1024, 256, 98320, stream>>>(Bq, w2q, b2, sc, Cbuf);
    kPw2<<<32768, 256, 0, stream>>>(Cbuf, sc, bn2g, bn2b, bn2m, bn2v, Dbuf, 8388608);
    kPw3<<<32768, 256, 0, stream>>>(Dbuf, sc, Ebuf, 8388608);
    // LSTM input projection (bias baked in), then persistent recurrent kernel
    kXproj<<<dim3(512, 4), 256, 65536, stream>>>(Ebuf, wiq, bih, bhh, xproj);
    kLstm<<<1, 1024, 196624, stream>>>(whq, xproj, accg, hnq);
    // multi-head quantized MLP
    kHeadGemm<<<dim3(4, 1, 4), 256, 65552, stream>>>(hnq, 0, hw0q, 32768, hb0, 256,
                                                     hpre, sc, 4, 256, 256, 128);
    kHeadPw<<<1024, 256, 0, stream>>>(hpre, sc, 4, 0, hbg, hbb, hbm, hbv, hyq);
    kHeadGemm<<<dim3(4, 1, 4), 256, 131088, stream>>>(hyq, 65536, hwhq, 131072, hbh, 512,
                                                      hpre, sc, 8, 256, 256, 256);
    kHeadPw<<<1024, 256, 0, stream>>>(hpre, sc, 8, 1, hbg, hbb, hbm, hbv, hyq);
    kHeadGemm<<<dim3(4, 1, 4), 256, 131088, stream>>>(hyq, 65536, hwhq + 65536, 131072,
                                                      hbh + 256, 512, hpre, sc, 12,
                                                      256, 256, 256);
    kHeadPw<<<1024, 256, 0, stream>>>(hpre, sc, 12, 2, hbg, hbb, hbm, hbv, hyq);
    kOut<<<1024, 256, 0, stream>>>(hyq, hwoq, hbo, (float*)d_out);
}
